// GraphNN_2422361555038
// MI455X (gfx1250) — matrix operand order repaired and checked by reading, not compiled, not run on hardware
//
#include <hip/hip_runtime.h>
#include <hip/hip_bf16.h>
#include <cstdint>
#include <cstddef>

// ---------------------------------------------------------------------------
// GAT (2 layers) + global mean pool + MLP for MI455X (gfx1250, wave32, WMMA).
// GEMMs run on v_wmma_f32_16x16x32_bf16 with register-resident B fragments
// and fully preloaded A fragments (16 WMMAs / wave, one load-wait per tile);
// edge aggregation is the bandwidth-bound part and uses one wave32 per edge
// with coalesced float4 atomics.
// ---------------------------------------------------------------------------

typedef __attribute__((ext_vector_type(16))) __bf16 v16bf;
typedef __attribute__((ext_vector_type(8)))  __bf16 v8bf;
typedef __attribute__((ext_vector_type(8)))  float  v8f;

#define DDIM 128
#define NEG_SLOPE 0.2f

// ---------------- helpers: order-preserving float<->uint for atomicMax -----
__device__ __forceinline__ unsigned f2ord(float f) {
    unsigned b = __float_as_uint(f);
    return (b & 0x80000000u) ? ~b : (b | 0x80000000u);
}
__device__ __forceinline__ float ord2f(unsigned e) {
    unsigned b = (e & 0x80000000u) ? (e & 0x7fffffffu) : ~e;
    return __uint_as_float(b);
}

// ---------------- f32 -> bf16 copy -----------------------------------------
__global__ void k_conv_bf16(const float* __restrict__ src, __bf16* __restrict__ dst,
                            int total) {
    int i = blockIdx.x * blockDim.x + threadIdx.x;
    if (i < total) dst[i] = (__bf16)src[i];
}

// ---------------- W[128x128] f32 row-major -> WT[n][k] bf16 ----------------
__global__ void k_transpose_w(const float* __restrict__ W, __bf16* __restrict__ WT) {
    int i = blockIdx.x * blockDim.x + threadIdx.x;   // i = n*128 + k
    int n = i >> 7, k = i & 127;
    WT[i] = (__bf16)W[k * DDIM + n];
}

// ---------------- WMMA GEMM:  H[N,128] = Xb[N,128] * W  --------------------
// Block = 256 thr (8 waves); block M-tile = 64 rows; wave w -> cols [16w,16w+16).
// B fragments (4 K-steps) loaded once into registers; A fragments for a
// subtile loaded in one clause, then 4 back-to-back WMMAs.
__global__ void k_gemm_wmma(const __bf16* __restrict__ Xb,
                            const __bf16* __restrict__ WT,
                            float* __restrict__ H, int nrows) {
    const int lane = threadIdx.x & 31;
    const int wave = threadIdx.x >> 5;          // 0..7
    const int n0   = wave * 16;
    const int half = lane >> 4;                 // 0 | 1 (lane group)
    const int l15  = lane & 15;
    const int m_base = blockIdx.x * 64;

    // B fragment (32x16 bf16 per K-step): lane col = lane&15,
    // lanes 0-15 K=k0..k0+7, k0+16..k0+23; lanes 16-31 shift K by 8 (matches A).
    const __bf16* brow = WT + (size_t)(n0 + l15) * DDIM;
    v16bf bf[4];
#pragma unroll
    for (int kk = 0; kk < 4; ++kk) {
        v8bf blo = *(const v8bf*)(brow + kk * 32 + half * 8);
        v8bf bhi = *(const v8bf*)(brow + kk * 32 + 16 + half * 8);
        bf[kk] = __builtin_shufflevector(blo, bhi,
                    0,1,2,3,4,5,6,7,8,9,10,11,12,13,14,15);
    }

#pragma unroll
    for (int sub = 0; sub < 4; ++sub) {
        const int m0 = m_base + sub * 16;
        if (m0 >= nrows) break;                 // uniform (scalar) branch
        int mr = m0 + l15;
        if (mr >= nrows) mr = nrows - 1;        // clamp (N%16==0 anyway)
        const __bf16* arow = Xb + (size_t)mr * DDIM;
        __builtin_prefetch(arow + 16 * DDIM, 0, 3);   // next M-subtile rows

        // Preload all 4 A fragments (8 x b128) -> one clause, one wait.
        // A fragment (16x32 bf16): lanes 0-15 K=k0..k0+7 (V0-3),
        // K=k0+16..k0+23 (V4-7); lanes 16-31 shift K by 8.
        v16bf a[4];
#pragma unroll
        for (int kk = 0; kk < 4; ++kk) {
            v8bf alo = *(const v8bf*)(arow + kk * 32 + half * 8);
            v8bf ahi = *(const v8bf*)(arow + kk * 32 + 16 + half * 8);
            a[kk] = __builtin_shufflevector(alo, ahi,
                        0,1,2,3,4,5,6,7,8,9,10,11,12,13,14,15);
        }

        v8f c = {};
#pragma unroll
        for (int kk = 0; kk < 4; ++kk)
            c = __builtin_amdgcn_wmma_f32_16x16x32_bf16(
                    false, a[kk], false, bf[kk], (short)0, c, false, false);

        // C/D layout: lane holds col n0+l15; VGPR r -> row m0+r (+8 if lane>=16)
        if (m0 + 16 <= nrows) {                 // fast path: no per-row guards
            float* hrow = H + (size_t)(m0 + half * 8) * DDIM + n0 + l15;
#pragma unroll
            for (int r = 0; r < 8; ++r) hrow[(size_t)r * DDIM] = c[r];
        } else {
#pragma unroll
            for (int r = 0; r < 8; ++r) {
                int m = m0 + r + half * 8;
                if (m < nrows) H[(size_t)m * DDIM + n0 + l15] = c[r];
            }
        }
    }
}

// ---------------- per-node attention logits: a = H[n] . att ----------------
__global__ void k_att(const float* __restrict__ H,
                      const float* __restrict__ atts, const float* __restrict__ attd,
                      float* __restrict__ as_, float* __restrict__ ad_, int N) {
    int t = blockIdx.x * blockDim.x + threadIdx.x;
    int n = t >> 5, lane = t & 31;
    if (n >= N) return;
    float4 h = *(const float4*)(H + (size_t)n * DDIM + lane * 4);
    float4 s = *(const float4*)(atts + lane * 4);
    float4 d = *(const float4*)(attd + lane * 4);
    float vs = h.x*s.x + h.y*s.y + h.z*s.z + h.w*s.w;
    float vd = h.x*d.x + h.y*d.y + h.z*d.z + h.w*d.w;
#pragma unroll
    for (int off = 16; off > 0; off >>= 1) {
        vs += __shfl_down(vs, off, 32);
        vd += __shfl_down(vd, off, 32);
    }
    if (lane == 0) { as_[n] = vs; ad_[n] = vd; }
}

// ---------------- init per-layer state -------------------------------------
__global__ void k_init_layer(float* __restrict__ acc, unsigned* __restrict__ menc,
                             float* __restrict__ ssum, int N) {
    int i = blockIdx.x * blockDim.x + threadIdx.x;
    if (i < N * DDIM) acc[i] = 0.0f;
    if (i < N) { menc[i] = f2ord(-__builtin_inff()); ssum[i] = 0.0f; }
}

// ---------------- edge pass A: alpha + segment max -------------------------
__global__ void k_edge_alpha(const int* __restrict__ ei, const float* __restrict__ as_,
                             const float* __restrict__ ad_, float* __restrict__ alpha,
                             unsigned* __restrict__ menc, int E, int N) {
    int e = blockIdx.x * blockDim.x + threadIdx.x;
    int tot = E + N;
    if (e >= tot) return;
    int s = (e < E) ? ei[e]     : (e - E);   // self loops appended after edges
    int d = (e < E) ? ei[E + e] : (e - E);
    float a = as_[s] + ad_[d];
    a = (a > 0.0f) ? a : NEG_SLOPE * a;      // leaky_relu
    alpha[e] = a;
    atomicMax(menc + d, f2ord(a));
}

// ---------------- edge pass B: exp + segment sum ---------------------------
__global__ void k_edge_exp(const int* __restrict__ ei, float* __restrict__ alpha,
                           const unsigned* __restrict__ menc, float* __restrict__ ssum,
                           int E, int N) {
    int e = blockIdx.x * blockDim.x + threadIdx.x;
    int tot = E + N;
    if (e >= tot) return;
    int d = (e < E) ? ei[E + e] : (e - E);
    float ev = __expf(alpha[e] - ord2f(menc[d]));
    alpha[e] = ev;
    atomicAdd(ssum + d, ev);
}

// ---------------- edge pass C: acc[dst] += coef * H[src]  (1 wave / edge) --
__global__ void k_edge_scatter(const int* __restrict__ ei, const float* __restrict__ alpha,
                               const float* __restrict__ ssum, const float* __restrict__ H,
                               float* __restrict__ acc, int E, int N) {
    int t = blockIdx.x * blockDim.x + threadIdx.x;
    int e = t >> 5, lane = t & 31;
    int tot = E + N;
    if (e >= tot) return;
    int s = (e < E) ? ei[e]     : (e - E);
    int d = (e < E) ? ei[E + e] : (e - E);
    float coef = alpha[e] / (ssum[d] + 1e-16f);
    float4 h = *(const float4*)(H + (size_t)s * DDIM + lane * 4);
    float* o = acc + (size_t)d * DDIM + lane * 4;
    atomicAdd(o + 0, coef * h.x);
    atomicAdd(o + 1, coef * h.y);
    atomicAdd(o + 2, coef * h.z);
    atomicAdd(o + 3, coef * h.w);
}

// ---------------- finalize: relu(acc + bias) -> f32 and/or bf16 ------------
__global__ void k_finalize(const float* __restrict__ acc, const float* __restrict__ bias,
                           float* __restrict__ outf, __bf16* __restrict__ outb, int total) {
    int i = blockIdx.x * blockDim.x + threadIdx.x;
    if (i >= total) return;
    float v = acc[i] + bias[i & (DDIM - 1)];
    v = (v > 0.0f) ? v : 0.0f;
    if (outf) outf[i] = v;
    if (outb) outb[i] = (__bf16)v;
}

// ---------------- pooling --------------------------------------------------
__global__ void k_pool_init(float* __restrict__ sums, float* __restrict__ cnt, int G) {
    int i = blockIdx.x * blockDim.x + threadIdx.x;
    if (i < G * DDIM) sums[i] = 0.0f;
    if (i < G) cnt[i] = 0.0f;
}

__global__ void k_pool(const float* __restrict__ Xf, const float* __restrict__ X0,
                       const int* __restrict__ batch, const int* __restrict__ flag,
                       float* __restrict__ sums, float* __restrict__ cnt, int N) {
    int t = blockIdx.x * blockDim.x + threadIdx.x;
    int n = t >> 5, lane = t & 31;
    if (n >= N) return;
    const float* feat = (*flag != 0) ? Xf : X0;   // use_message_passing select
    int g = batch[n];
    float4 v = *(const float4*)(feat + (size_t)n * DDIM + lane * 4);
    float* o = sums + (size_t)g * DDIM + lane * 4;
    atomicAdd(o + 0, v.x);
    atomicAdd(o + 1, v.y);
    atomicAdd(o + 2, v.z);
    atomicAdd(o + 3, v.w);
    if (lane == 0) atomicAdd(cnt + g, 1.0f);
}

// ---------------- final MLP: one block (128 thr) per graph -----------------
__global__ void k_mlp(const float* __restrict__ sums, const float* __restrict__ cnt,
                      const float* __restrict__ W1, const float* __restrict__ b1,
                      const float* __restrict__ W2, const float* __restrict__ b2,
                      float* __restrict__ out) {
    __shared__ float gm[DDIM];
    __shared__ float red[DDIM];
    int g = blockIdx.x, j = threadIdx.x;
    float c = cnt[g];
    c = (c > 1.0f) ? c : 1.0f;
    gm[j] = sums[(size_t)g * DDIM + j] / c;
    __syncthreads();
    float h = b1[j];
#pragma unroll 8
    for (int k = 0; k < DDIM; ++k) h += gm[k] * W1[k * DDIM + j];
    h = (h > 0.0f) ? h : 0.0f;
    red[j] = h * W2[j];
    __syncthreads();
    for (int st = 64; st > 0; st >>= 1) {
        if (j < st) red[j] += red[j + st];
        __syncthreads();
    }
    if (j == 0) out[g] = red[0] + b2[0];
}

// ===========================================================================
extern "C" void kernel_launch(void* const* d_in, const int* in_sizes, int n_in,
                              void* d_out, int out_size, void* d_ws, size_t ws_size,
                              hipStream_t stream) {
    const float* x     = (const float*)d_in[0];
    const int*   ei    = (const int*)d_in[1];
    const int*   batch = (const int*)d_in[3];
    const float* Wg1   = (const float*)d_in[4];
    const float* as1   = (const float*)d_in[5];
    const float* ad1   = (const float*)d_in[6];
    const float* bg1   = (const float*)d_in[7];
    const float* Wg2   = (const float*)d_in[8];
    const float* as2   = (const float*)d_in[9];
    const float* ad2   = (const float*)d_in[10];
    const float* bg2   = (const float*)d_in[11];
    const float* Wl1   = (const float*)d_in[12];
    const float* bl1   = (const float*)d_in[13];
    const float* Wl2   = (const float*)d_in[14];
    const float* bl2   = (const float*)d_in[15];
    const int*   ump   = (const int*)d_in[17];    // use_message_passing (device)

    const int N = in_sizes[0] / DDIM;             // 50000
    const int E = in_sizes[1] / 2;                // 800000
    const int G = out_size;                       // 64
    const int TOT = E + N;
    const int NF = N * DDIM;

    // ---- workspace layout (256B aligned slabs) ----
    char* wsb = (char*)d_ws;
    size_t off = 0;
    auto alloc = [&](size_t bytes) -> char* {
        char* p = wsb + off;
        off = (off + bytes + 255) & ~(size_t)255;
        return p;
    };
    __bf16*   Xb    = (__bf16*)alloc((size_t)NF * 2);        // bf16 GEMM input
    __bf16*   WT    = (__bf16*)alloc((size_t)DDIM*DDIM * 2); // bf16 W^T
    float*    H     = (float*)alloc((size_t)NF * 4);         // GEMM out / final X
    float*    acc   = (float*)alloc((size_t)NF * 4);         // message acc
    float*    alpha = (float*)alloc((size_t)TOT * 4);
    float*    asrc  = (float*)alloc((size_t)N * 4);
    float*    adst  = (float*)alloc((size_t)N * 4);
    unsigned* menc  = (unsigned*)alloc((size_t)N * 4);
    float*    ssum  = (float*)alloc((size_t)N * 4);
    float*    sums  = (float*)alloc((size_t)G * DDIM * 4);
    float*    cnt   = (float*)alloc((size_t)G * 4);
    (void)ws_size; (void)n_in;

    const int T = 256;
    dim3 b(T);
    dim3 gNF((NF + T - 1) / T);
    dim3 gW((DDIM * DDIM + T - 1) / T);
    dim3 gGemm((N + 63) / 64);                       // 64-row M-tiles
    dim3 gNode(((size_t)N * 32 + T - 1) / T);        // 1 wave / node
    dim3 gEdge((TOT + T - 1) / T);
    dim3 gEdgeW(((size_t)TOT * 32 + T - 1) / T);     // 1 wave / edge
    dim3 gPoolI((G * DDIM + T - 1) / T);

    // ---- convert x to bf16 ----
    k_conv_bf16<<<gNF, b, 0, stream>>>(x, Xb, NF);

    // =================== GAT layer 1 ===================
    k_transpose_w<<<gW, b, 0, stream>>>(Wg1, WT);
    k_gemm_wmma<<<gGemm, b, 0, stream>>>(Xb, WT, H, N);
    k_att<<<gNode, b, 0, stream>>>(H, as1, ad1, asrc, adst, N);
    k_init_layer<<<gNF, b, 0, stream>>>(acc, menc, ssum, N);
    k_edge_alpha<<<gEdge, b, 0, stream>>>(ei, asrc, adst, alpha, menc, E, N);
    k_edge_exp<<<gEdge, b, 0, stream>>>(ei, alpha, menc, ssum, E, N);
    k_edge_scatter<<<gEdgeW, b, 0, stream>>>(ei, alpha, ssum, H, acc, E, N);
    // relu(acc+bias) -> Xb (bf16 input to layer 2); f32 copy not needed here
    k_finalize<<<gNF, b, 0, stream>>>(acc, bg1, (float*)nullptr, Xb, NF);

    // =================== GAT layer 2 ===================
    k_transpose_w<<<gW, b, 0, stream>>>(Wg2, WT);
    k_gemm_wmma<<<gGemm, b, 0, stream>>>(Xb, WT, H, N);
    k_att<<<gNode, b, 0, stream>>>(H, as2, ad2, asrc, adst, N);
    k_init_layer<<<gNF, b, 0, stream>>>(acc, menc, ssum, N);
    k_edge_alpha<<<gEdge, b, 0, stream>>>(ei, asrc, adst, alpha, menc, E, N);
    k_edge_exp<<<gEdge, b, 0, stream>>>(ei, alpha, menc, ssum, E, N);
    k_edge_scatter<<<gEdgeW, b, 0, stream>>>(ei, alpha, ssum, H, acc, E, N);
    // relu(acc+bias) -> H (f32 final node features); H no longer read
    k_finalize<<<gNF, b, 0, stream>>>(acc, bg2, H, (__bf16*)nullptr, NF);

    // =================== pool + MLP ===================
    k_pool_init<<<gPoolI, b, 0, stream>>>(sums, cnt, G);
    k_pool<<<gNode, b, 0, stream>>>(H, x, batch, ump, sums, cnt, N);
    k_mlp<<<dim3(G), dim3(DDIM), 0, stream>>>(sums, cnt, Wl1, bl1, Wl2, bl2,
                                              (float*)d_out);
}
